// EnhancedGraphSAGEModel_85306640433357
// MI455X (gfx1250) — matrix operand order repaired and checked by reading, not compiled, not run on hardware
//
#include <hip/hip_runtime.h>
#include <math.h>

#define N_NODES 100000
#define N_EDGES 800000
#define IN_CH   128
#define HID     512
#define BN_EPS  1e-5f

typedef __attribute__((ext_vector_type(16))) __bf16      v16bf;
typedef __attribute__((ext_vector_type(8)))  float       v8f;
typedef __attribute__((ext_vector_type(4)))  unsigned int u32x4;
typedef __attribute__((ext_vector_type(4)))  float       f32x4;
typedef __attribute__((ext_vector_type(4)))  __bf16      v4bf;

union Frag16 { v16bf v; u32x4 u[2]; };

// ---------------------------------------------------------------------------
// Utility kernels
// ---------------------------------------------------------------------------
__global__ void zero_f32(float* __restrict__ p, long long n) {
    long long i = (long long)blockIdx.x * blockDim.x + threadIdx.x;
    long long stride = (long long)gridDim.x * blockDim.x;
    for (; i < n; i += stride) p[i] = 0.0f;
}

__global__ void deg_kernel(const int* __restrict__ dst, float* __restrict__ deg) {
    int e = blockIdx.x * blockDim.x + threadIdx.x;
    if (e < N_EDGES) atomicAdd(&deg[dst[e]], 1.0f);
}

__global__ void invdeg_kernel(const float* __restrict__ deg, float* __restrict__ inv) {
    int i = blockIdx.x * blockDim.x + threadIdx.x;
    if (i < N_NODES) inv[i] = 1.0f / fmaxf(deg[i], 1.0f);
}

// scatter: agg[dst] += feat[src], 4 floats per thread
__global__ void scatter_add(const float* __restrict__ feat, const int* __restrict__ src,
                            const int* __restrict__ dst, float* __restrict__ agg, int d4) {
    long long idx = (long long)blockIdx.x * blockDim.x + threadIdx.x;
    long long total = (long long)N_EDGES * d4;
    if (idx >= total) return;
    int e = (int)(idx / d4);
    int c = (int)(idx % d4) * 4;
    int d = d4 * 4;
    int s = src[e], dn = dst[e];
    f32x4 v = *(const f32x4*)(feat + (size_t)s * d + c);
    float* ap = agg + (size_t)dn * d + c;
    atomicAdd(ap + 0, v.x);
    atomicAdd(ap + 1, v.y);
    atomicAdd(ap + 2, v.z);
    atomicAdd(ap + 3, v.w);
}

__device__ __forceinline__ v4bf cvt4(f32x4 v) {
    v4bf r;
    r.x = (__bf16)v.x; r.y = (__bf16)v.y; r.z = (__bf16)v.z; r.w = (__bf16)v.w;
    return r;
}

// Build combined bf16 A matrix: [agg*invdeg | x], row stride 2*d (4 feats/thread)
__global__ void pack_conv(const float* __restrict__ agg, const float* __restrict__ invdeg,
                          const float* __restrict__ x, __bf16* __restrict__ out, int d) {
    int d4 = d >> 2;
    long long idx = (long long)blockIdx.x * blockDim.x + threadIdx.x;
    long long total = (long long)N_NODES * d4;
    if (idx >= total) return;
    int m = (int)(idx / d4);
    int f = (int)(idx % d4) * 4;
    size_t ib = (size_t)m * d + f;
    f32x4 av = *(const f32x4*)(agg + ib);
    f32x4 xv = *(const f32x4*)(x + ib);
    float inv = invdeg[m];
    av.x *= inv; av.y *= inv; av.z *= inv; av.w *= inv;
    size_t ob = (size_t)m * (size_t)(2 * d);
    *(v4bf*)(out + ob + f)     = cvt4(av);
    *(v4bf*)(out + ob + d + f) = cvt4(xv);
}

__global__ void pack_cls(const float* __restrict__ h, __bf16* __restrict__ out, long long n4) {
    long long i = (long long)blockIdx.x * blockDim.x + threadIdx.x;
    if (i >= n4) return;
    f32x4 v = *(const f32x4*)(h + i * 4);
    *(v4bf*)(out + i * 4) = cvt4(v);
}

// Transpose+convert weight W[k][n] (Kpart x Nw, row major) into bf16 Wt[n][kOff+k]
// where Wt has row stride Kfull.
__global__ void wt_pack(const float* __restrict__ W, __bf16* __restrict__ out,
                        int Kpart, int Nw, int Kfull, int kOff) {
    int idx = blockIdx.x * blockDim.x + threadIdx.x;
    if (idx >= Kpart * Nw) return;
    int kk = idx / Nw;
    int n  = idx % Nw;
    out[(size_t)n * Kfull + kOff + kk] = (__bf16)W[idx];
}

// Fold BN to y = v*s + t
__global__ void bn_prep(const float* __restrict__ g, const float* __restrict__ be,
                        const float* __restrict__ mn, const float* __restrict__ vr,
                        float* __restrict__ s, float* __restrict__ t, int n) {
    int i = blockIdx.x * blockDim.x + threadIdx.x;
    if (i >= n) return;
    float ss = g[i] * rsqrtf(vr[i] + BN_EPS);
    s[i] = ss;
    t[i] = be[i] - mn[i] * ss;
}

// ---------------------------------------------------------------------------
// bf16 WMMA GEMM: out[M x N] = A[M x K](bf16) * Wt[N x K]^T(bf16) + epilogue
// mode 0: val = relu((acc+bias)*s + t) (+ resid)      [conv layers]
// mode 1: val = relu(acc+bias)*s + t                  [classifier]
// Wave computes a 32x64 tile (2 A-frags x 4 B-frags = 8 WMMA / K-step);
// every B fragment is reused for two WMMAs -> 1.5 loads per WMMA.
// block = 4 waves (128 threads); grid.x = M/32, grid.y = ceil(N/256)
// ---------------------------------------------------------------------------
__device__ __forceinline__ void gemm_epilogue(v8f c, int n, int mbase, int N, int mode,
                                              const float* __restrict__ bias,
                                              const float* __restrict__ s,
                                              const float* __restrict__ t,
                                              const float* __restrict__ resid,
                                              float* __restrict__ out) {
    float bb = bias[n];
    float ss = s[n];
    float tt = t[n];
#pragma unroll
    for (int v = 0; v < 8; ++v) {
        int m = mbase + v;
        float acc = c[v];
        float val;
        if (mode == 0) {
            val = fmaxf((acc + bb) * ss + tt, 0.0f);
            if (resid) val += resid[(size_t)m * N + n];
        } else {
            val = fmaxf(acc + bb, 0.0f) * ss + tt;
        }
        out[(size_t)m * N + n] = val;
    }
}

__global__ void gemm_wmma_bf16(const __bf16* __restrict__ A, const __bf16* __restrict__ Wt,
                               const float* __restrict__ bias, const float* __restrict__ s,
                               const float* __restrict__ t, const float* __restrict__ resid,
                               float* __restrict__ out, int K, int N, int mode) {
    const int lane   = threadIdx.x & 31;
    const int wave   = threadIdx.x >> 5;
    const int lane16 = lane & 15;
    const int half   = lane >> 4;
    const int m0 = blockIdx.x << 5;            // 32 rows per wave-tile
    const int n0 = blockIdx.y * 256 + wave * 64;
    if (n0 >= N) return;

    v8f c00 = {}, c01 = {}, c02 = {}, c03 = {};   // rows m0..m0+15
    v8f c10 = {}, c11 = {}, c12 = {}, c13 = {};   // rows m0+16..m0+31

    // A fragment: row = m + lane16; 16 consecutive k starting at kb + half*16
    const __bf16* Arow0 = A + (size_t)(m0 + lane16) * K + half * 16;
    const __bf16* Arow1 = Arow0 + (size_t)16 * K;
    // B fragment: col = n + lane16; 16 consecutive k starting at kb + half*16
    const __bf16* B0 = Wt + (size_t)(n0 +  0 + lane16) * K + half * 16;
    const __bf16* B1 = Wt + (size_t)(n0 + 16 + lane16) * K + half * 16;
    const __bf16* B2 = Wt + (size_t)(n0 + 32 + lane16) * K + half * 16;
    const __bf16* B3 = Wt + (size_t)(n0 + 48 + lane16) * K + half * 16;

    for (int kb = 0; kb < K; kb += 32) {
        Frag16 a0, a1, b;
        a0.u[0] = *(const u32x4*)(Arow0 + kb);
        a0.u[1] = *(const u32x4*)(Arow0 + kb + 8);
        a1.u[0] = *(const u32x4*)(Arow1 + kb);
        a1.u[1] = *(const u32x4*)(Arow1 + kb + 8);

#define WMMA_PAIR(Bp, cA, cB)                                                              \
        b.u[0] = *(const u32x4*)(Bp + kb);                                                 \
        b.u[1] = *(const u32x4*)(Bp + kb + 8);                                             \
        cA = __builtin_amdgcn_wmma_f32_16x16x32_bf16(false, a0.v, false, b.v, (short)0, cA, false, false); \
        cB = __builtin_amdgcn_wmma_f32_16x16x32_bf16(false, a1.v, false, b.v, (short)0, cB, false, false);

        WMMA_PAIR(B0, c00, c10)
        WMMA_PAIR(B1, c01, c11)
        WMMA_PAIR(B2, c02, c12)
        WMMA_PAIR(B3, c03, c13)
#undef WMMA_PAIR
    }

    const int mb0 = m0 + half * 8;
    const int mb1 = m0 + 16 + half * 8;
    gemm_epilogue(c00, n0 +  0 + lane16, mb0, N, mode, bias, s, t, resid, out);
    gemm_epilogue(c01, n0 + 16 + lane16, mb0, N, mode, bias, s, t, resid, out);
    gemm_epilogue(c02, n0 + 32 + lane16, mb0, N, mode, bias, s, t, resid, out);
    gemm_epilogue(c03, n0 + 48 + lane16, mb0, N, mode, bias, s, t, resid, out);
    gemm_epilogue(c10, n0 +  0 + lane16, mb1, N, mode, bias, s, t, resid, out);
    gemm_epilogue(c11, n0 + 16 + lane16, mb1, N, mode, bias, s, t, resid, out);
    gemm_epilogue(c12, n0 + 32 + lane16, mb1, N, mode, bias, s, t, resid, out);
    gemm_epilogue(c13, n0 + 48 + lane16, mb1, N, mode, bias, s, t, resid, out);
}

// Final 128 -> 1 layer (f32 GEMV, wave per node)
__global__ void final_gemv(const float* __restrict__ H, const float* __restrict__ W,
                           const float* __restrict__ b, float* __restrict__ out) {
    int gw   = (blockIdx.x * blockDim.x + threadIdx.x) >> 5;
    int lane = threadIdx.x & 31;
    if (gw >= N_NODES) return;
    const float* row = H + (size_t)gw * 128;
    float sum = 0.0f;
#pragma unroll
    for (int k = lane; k < 128; k += 32) sum += row[k] * W[k];
#pragma unroll
    for (int off = 16; off; off >>= 1) sum += __shfl_down(sum, off, 32);
    if (lane == 0) out[gw] = sum + b[0];
}

// ---------------------------------------------------------------------------
// Host-side launch
// ---------------------------------------------------------------------------
extern "C" void kernel_launch(void* const* d_in, const int* in_sizes, int n_in,
                              void* d_out, int out_size, void* d_ws, size_t ws_size,
                              hipStream_t stream) {
    (void)in_sizes; (void)n_in; (void)out_size; (void)ws_size;

    // ---- inputs (setup_inputs dict order, depth-first flatten) ----
    const float* x    = (const float*)d_in[0];
    const int*   edge = (const int*)d_in[1];
    const int*   esrc = edge;
    const int*   edst = edge + N_EDGES;

    const float* Wl[3]   = {(const float*)d_in[2],  (const float*)d_in[5],  (const float*)d_in[8]};
    const float* blv[3]  = {(const float*)d_in[3],  (const float*)d_in[6],  (const float*)d_in[9]};
    const float* Wr[3]   = {(const float*)d_in[4],  (const float*)d_in[7],  (const float*)d_in[10]};
    const float* bng[3]  = {(const float*)d_in[11], (const float*)d_in[15], (const float*)d_in[19]};
    const float* bnb[3]  = {(const float*)d_in[12], (const float*)d_in[16], (const float*)d_in[20]};
    const float* bnm[3]  = {(const float*)d_in[13], (const float*)d_in[17], (const float*)d_in[21]};
    const float* bnv[3]  = {(const float*)d_in[14], (const float*)d_in[18], (const float*)d_in[22]};

    const float* cW[4]   = {(const float*)d_in[23], (const float*)d_in[29], (const float*)d_in[35], (const float*)d_in[41]};
    const float* cb[4]   = {(const float*)d_in[24], (const float*)d_in[30], (const float*)d_in[36], (const float*)d_in[42]};
    const float* cg[3]   = {(const float*)d_in[25], (const float*)d_in[31], (const float*)d_in[37]};
    const float* cbe[3]  = {(const float*)d_in[26], (const float*)d_in[32], (const float*)d_in[38]};
    const float* cm[3]   = {(const float*)d_in[27], (const float*)d_in[33], (const float*)d_in[39]};
    const float* cv[3]   = {(const float*)d_in[28], (const float*)d_in[34], (const float*)d_in[40]};

    float* outp = (float*)d_out;

    // ---- workspace carve ----
    char* w = (char*)d_ws;
    auto alloc = [&](size_t bytes) -> void* {
        void* p = (void*)w;
        w += (bytes + 255) & ~(size_t)255;
        return p;
    };
    float*  hA     = (float*)alloc((size_t)N_NODES * HID * 4);        // activations
    float*  agg    = (float*)alloc((size_t)N_NODES * HID * 4);        // scatter target
    __bf16* Acomb  = (__bf16*)alloc((size_t)N_NODES * 2 * HID * 2);   // bf16 A matrix
    float*  deg    = (float*)alloc((size_t)N_NODES * 4);
    float*  invdeg = (float*)alloc((size_t)N_NODES * 4);
    __bf16* WtConv[3];
    WtConv[0] = (__bf16*)alloc((size_t)512 * 256  * 2);
    WtConv[1] = (__bf16*)alloc((size_t)512 * 1024 * 2);
    WtConv[2] = (__bf16*)alloc((size_t)512 * 1024 * 2);
    __bf16* WtCls[3];
    WtCls[0] = (__bf16*)alloc((size_t)512 * 512 * 2);
    WtCls[1] = (__bf16*)alloc((size_t)256 * 512 * 2);
    WtCls[2] = (__bf16*)alloc((size_t)128 * 256 * 2);
    float* sConv[3], *tConv[3], *sCls[3], *tCls[3];
    for (int l = 0; l < 3; ++l) { sConv[l] = (float*)alloc(512 * 4); tConv[l] = (float*)alloc(512 * 4); }
    int clsN[4] = {512, 256, 128, 1};
    int clsK[4] = {512, 512, 256, 128};
    for (int l = 0; l < 3; ++l) { sCls[l] = (float*)alloc(clsN[l] * 4); tCls[l] = (float*)alloc(clsN[l] * 4); }

    const int TB = 256;

    // ---- weight / BN prep (small, once per call) ----
    for (int l = 0; l < 3; ++l) {
        int dl = (l == 0) ? IN_CH : HID;
        int n1 = dl * 512;
        hipLaunchKernelGGL(wt_pack, dim3((n1 + TB - 1) / TB), dim3(TB), 0, stream,
                           Wl[l], WtConv[l], dl, 512, 2 * dl, 0);
        hipLaunchKernelGGL(wt_pack, dim3((n1 + TB - 1) / TB), dim3(TB), 0, stream,
                           Wr[l], WtConv[l], dl, 512, 2 * dl, dl);
        hipLaunchKernelGGL(bn_prep, dim3((512 + TB - 1) / TB), dim3(TB), 0, stream,
                           bng[l], bnb[l], bnm[l], bnv[l], sConv[l], tConv[l], 512);
    }
    for (int l = 0; l < 3; ++l) {
        int n1 = clsK[l] * clsN[l];
        hipLaunchKernelGGL(wt_pack, dim3((n1 + TB - 1) / TB), dim3(TB), 0, stream,
                           cW[l], WtCls[l], clsK[l], clsN[l], clsK[l], 0);
        hipLaunchKernelGGL(bn_prep, dim3((clsN[l] + TB - 1) / TB), dim3(TB), 0, stream,
                           cg[l], cbe[l], cm[l], cv[l], sCls[l], tCls[l], clsN[l]);
    }

    // ---- degree (same for every layer) ----
    hipLaunchKernelGGL(zero_f32, dim3(512), dim3(TB), 0, stream, deg, (long long)N_NODES);
    hipLaunchKernelGGL(deg_kernel, dim3((N_EDGES + TB - 1) / TB), dim3(TB), 0, stream, edst, deg);
    hipLaunchKernelGGL(invdeg_kernel, dim3((N_NODES + TB - 1) / TB), dim3(TB), 0, stream, deg, invdeg);

    // ---- conv layers ----
    for (int l = 0; l < 3; ++l) {
        int dl = (l == 0) ? IN_CH : HID;
        const float* feat = (l == 0) ? x : hA;
        long long aggN = (long long)N_NODES * dl;
        hipLaunchKernelGGL(zero_f32, dim3(4096), dim3(TB), 0, stream, agg, aggN);

        int d4 = dl / 4;
        long long scatN = (long long)N_EDGES * d4;
        hipLaunchKernelGGL(scatter_add, dim3((unsigned)((scatN + TB - 1) / TB)), dim3(TB), 0, stream,
                           feat, esrc, edst, agg, d4);

        long long packN = (long long)N_NODES * d4;
        hipLaunchKernelGGL(pack_conv, dim3((unsigned)((packN + TB - 1) / TB)), dim3(TB), 0, stream,
                           agg, invdeg, feat, Acomb, dl);

        const float* resid = (l == 0) ? (const float*)nullptr : hA;
        hipLaunchKernelGGL(gemm_wmma_bf16, dim3(N_NODES / 32, 2), dim3(128), 0, stream,
                           Acomb, WtConv[l], blv[l], sConv[l], tConv[l], resid, hA,
                           2 * dl, 512, 0);
    }

    // ---- classifier layers 0..2 (GEMM), 3 (GEMV) ----
    for (int l = 0; l < 3; ++l) {
        long long nel4 = (long long)N_NODES * clsK[l] / 4;
        hipLaunchKernelGGL(pack_cls, dim3((unsigned)((nel4 + TB - 1) / TB)), dim3(TB), 0, stream,
                           hA, Acomb, nel4);
        int gy = (clsN[l] + 255) / 256;
        hipLaunchKernelGGL(gemm_wmma_bf16, dim3(N_NODES / 32, gy), dim3(128), 0, stream,
                           Acomb, WtCls[l], cb[l], sCls[l], tCls[l], (const float*)nullptr, hA,
                           clsK[l], clsN[l], 1);
    }

    hipLaunchKernelGGL(final_gemv, dim3((N_NODES * 32 + TB - 1) / TB), dim3(TB), 0, stream,
                       hA, cW[3], cb[3], outp);
}
